// SAGEConvWithMultipleLinearLayers_88433376624936
// MI455X (gfx1250) — matrix operand order repaired and checked by reading, not compiled, not run on hardware
//
#include <hip/hip_runtime.h>
#include <stdint.h>

// ---------------------------------------------------------------------------
// Types / helpers
// ---------------------------------------------------------------------------
typedef __attribute__((ext_vector_type(16))) __bf16 v16bf;
typedef __attribute__((ext_vector_type(8)))  float  v8f;
typedef __attribute__((ext_vector_type(4)))  int    v4i;

#define AS1 __attribute__((address_space(1)))
#define AS3 __attribute__((address_space(3)))

#if defined(__has_builtin)
#if __has_builtin(__builtin_amdgcn_global_load_async_to_lds_b128) && \
    __has_builtin(__builtin_amdgcn_s_wait_asynccnt)
#define USE_ASYNC_LDS 1
#endif
#endif
#ifndef USE_ASYNC_LDS
#define USE_ASYNC_LDS 0
#endif

__device__ __forceinline__ unsigned short f32_to_bf16(float f) {
  unsigned u = __builtin_bit_cast(unsigned, f);
  u += 0x7FFFu + ((u >> 16) & 1u);          // round-to-nearest-even
  return (unsigned short)(u >> 16);
}
__device__ __forceinline__ float bf16_to_f32(unsigned short h) {
  unsigned u = ((unsigned)h) << 16;
  return __builtin_bit_cast(float, u);
}

// 16B global -> LDS copy: async (ASYNCcnt-tracked) when available, sync else.
__device__ __forceinline__ void copy_b128(const unsigned short* g,
                                          unsigned short* l) {
#if USE_ASYNC_LDS
  __builtin_amdgcn_global_load_async_to_lds_b128(
      (AS1 v4i*)const_cast<unsigned short*>(g), (AS3 v4i*)l, 0, 0);
#else
  *(uint4*)l = *(const uint4*)g;
#endif
}
__device__ __forceinline__ void wait_async0() {
#if USE_ASYNC_LDS
  __builtin_amdgcn_s_wait_asynccnt(0);
#endif
}

// ---------------------------------------------------------------------------
// f32 -> bf16 conversion (weights / activations)
// ---------------------------------------------------------------------------
__global__ void f2bf_kernel(const float* __restrict__ in,
                            unsigned short* __restrict__ out, long long n) {
  long long t = (long long)blockIdx.x * blockDim.x + threadIdx.x;
  if (t < n) out[t] = f32_to_bf16(in[t]);
}

// ---------------------------------------------------------------------------
// Graph aggregation: degree, scatter-add, mean+convert
// ---------------------------------------------------------------------------
__global__ void degree_kernel(const int* __restrict__ dst,
                              float* __restrict__ deg, int E) {
  int e = blockIdx.x * blockDim.x + threadIdx.x;
  if (e < E) atomicAdd(&deg[dst[e]], 1.0f);
}

__global__ void scatter_f32_kernel(const float* __restrict__ x,
                                   const int* __restrict__ src,
                                   const int* __restrict__ dst,
                                   float* __restrict__ agg,
                                   int E, int F, int chunkShift) {
  long long t = (long long)blockIdx.x * blockDim.x + threadIdx.x;
  long long total = (long long)E << chunkShift;
  if (t >= total) return;
  int e = (int)(t >> chunkShift);
  int c = (int)(t & ((1 << chunkShift) - 1)) * 4;
  int s = src[e], d = dst[e];
  const float4 v = *(const float4*)(x + (size_t)s * F + c);
  float* p = agg + (size_t)d * F + c;
  atomicAdd(p + 0, v.x); atomicAdd(p + 1, v.y);
  atomicAdd(p + 2, v.z); atomicAdd(p + 3, v.w);
}

__global__ void scatter_bf16_kernel(const unsigned short* __restrict__ h,
                                    const int* __restrict__ src,
                                    const int* __restrict__ dst,
                                    float* __restrict__ agg,
                                    int E, int F, int chunkShift) {
  long long t = (long long)blockIdx.x * blockDim.x + threadIdx.x;
  long long total = (long long)E << chunkShift;
  if (t >= total) return;
  int e = (int)(t >> chunkShift);
  int c = (int)(t & ((1 << chunkShift) - 1)) * 4;
  int s = src[e], d = dst[e];
  const uint2 v = *(const uint2*)(h + (size_t)s * F + c);
  float* p = agg + (size_t)d * F + c;
  atomicAdd(p + 0, bf16_to_f32((unsigned short)(v.x & 0xffffu)));
  atomicAdd(p + 1, bf16_to_f32((unsigned short)(v.x >> 16)));
  atomicAdd(p + 2, bf16_to_f32((unsigned short)(v.y & 0xffffu)));
  atomicAdd(p + 3, bf16_to_f32((unsigned short)(v.y >> 16)));
}

__global__ void mean_bf16_kernel(const float* __restrict__ agg,
                                 const float* __restrict__ deg,
                                 unsigned short* __restrict__ out,
                                 long long total, int fShift) {
  long long t = (long long)blockIdx.x * blockDim.x + threadIdx.x;
  if (t >= total) return;
  int i = (int)(t >> fShift);
  float dg = deg[i];
  float r = (dg > 1.0f) ? (1.0f / dg) : 1.0f;   // mean with max(deg,1)
  out[t] = f32_to_bf16(agg[t] * r);
}

// ---------------------------------------------------------------------------
// Fused bf16 WMMA GEMM:  out = act( A1 @ W1 [+ A2 @ W2] + bias )
// A* : [M,K] bf16 row-major, W* : [K,N] bf16 row-major, out: [M,N] bf16
// Block = 256 threads (8 waves). Block tile = 64(M) x 64(N). K-step = 32.
// Wave w: N slice = (w&3)*16; M tiles = {(w>>2)*2, (w>>2)*2+1} (two accums,
// B fragment reused). Double-buffered LDS, async global->LDS pipeline.
// ---------------------------------------------------------------------------
__global__ __launch_bounds__(256) void gemm_bf16_wmma(
    const unsigned short* __restrict__ A1,
    const unsigned short* __restrict__ A2,   // nullptr => single input
    const unsigned short* __restrict__ W1,
    const unsigned short* __restrict__ W2,
    const float* __restrict__ bias,
    unsigned short* __restrict__ out,
    int M, int K, int N, int do_relu) {
  constexpr int BSTR = 72;                    // B-tile LDS row stride (halfs)
  __shared__ unsigned short As1[2][64 * 32];
  __shared__ unsigned short As2[2][64 * 32];
  __shared__ unsigned short Bs1[2][32 * BSTR];
  __shared__ unsigned short Bs2[2][32 * BSTR];

  const int lane = threadIdx.x & 31;
  const int wave = threadIdx.x >> 5;          // 0..7
  const int ni   = wave & 3;                  // N sub-tile
  const int mi0  = (wave >> 2) * 2;           // first of two M sub-tiles
  const int m0 = blockIdx.y * 64;
  const int n0 = blockIdx.x * 64;
  const bool dual = (A2 != nullptr);

  v8f acc0, acc1;
#pragma unroll
  for (int i = 0; i < 8; ++i) { acc0[i] = 0.0f; acc1[i] = 0.0f; }

  // stage K-step ks into LDS buffer buf (one b128 per thread per matrix)
  auto stage = [&](int ks, int buf) {
    const int k0 = ks * 32;
    {
      const int r  = threadIdx.x >> 2;        // 0..63 (M row in tile)
      const int cc = (threadIdx.x & 3) * 8;   // 0,8,16,24 (K col)
      const int gr = (m0 + r < M) ? (m0 + r) : (M - 1);
      const size_t ga = (size_t)gr * K + k0 + cc;
      copy_b128(A1 + ga, &As1[buf][r * 32 + cc]);
      if (dual) copy_b128(A2 + ga, &As2[buf][r * 32 + cc]);
    }
    {
      const int r  = threadIdx.x >> 3;        // 0..31 (K row)
      const int cc = (threadIdx.x & 7) * 8;   // 0..56 (N col)
      const size_t gb = (size_t)(k0 + r) * N + n0 + cc;
      copy_b128(W1 + gb, &Bs1[buf][r * BSTR + cc]);
      if (dual) copy_b128(W2 + gb, &Bs2[buf][r * BSTR + cc]);
    }
  };

  const int KSTEPS = K >> 5;
  stage(0, 0);

  const int mloc = lane & 15;
  const int lhi  = lane >> 4;

  for (int s = 0; s < KSTEPS; ++s) {
    const int buf = s & 1;
    wait_async0();        // stage-s data landed (async loads complete in order)
    __syncthreads();      // stage-s visible to all; step s-1 reads finished
    if (s + 1 < KSTEPS) stage(s + 1, buf ^ 1);

    // B fragment (32x16 tile): lane L holds col N=L&15; half j holds K = j + 8*((L>>4)+(j>>3))
    v16bf b1f;
#pragma unroll
    for (int j = 0; j < 16; ++j) {
      const int kk = j + 8 * (lhi + (j >> 3));
      b1f[j] = __builtin_bit_cast(__bf16, Bs1[buf][kk * BSTR + ni * 16 + mloc]);
    }
    // A fragments (ISA 7.12.2, 16-bit A 16x32):
    //   lane L holds row M=L&15; half j holds K = j + 8*((L>>4)+(j>>3))
    v16bf a0f, a1f;
#pragma unroll
    for (int j = 0; j < 16; ++j) {
      const int kk = j + 8 * (lhi + (j >> 3));
      a0f[j] = __builtin_bit_cast(__bf16, As1[buf][(mi0 * 16 + mloc) * 32 + kk]);
      a1f[j] = __builtin_bit_cast(__bf16, As1[buf][((mi0 + 1) * 16 + mloc) * 32 + kk]);
    }
    acc0 = __builtin_amdgcn_wmma_f32_16x16x32_bf16(false, a0f, false, b1f,
                                                   (short)0, acc0, false, false);
    acc1 = __builtin_amdgcn_wmma_f32_16x16x32_bf16(false, a1f, false, b1f,
                                                   (short)0, acc1, false, false);
    if (dual) {
      v16bf b2f, c0f, c1f;
#pragma unroll
      for (int j = 0; j < 16; ++j) {
        const int kk = j + 8 * (lhi + (j >> 3));
        b2f[j] = __builtin_bit_cast(__bf16, Bs2[buf][kk * BSTR + ni * 16 + mloc]);
      }
#pragma unroll
      for (int j = 0; j < 16; ++j) {
        const int kk = j + 8 * (lhi + (j >> 3));
        c0f[j] = __builtin_bit_cast(__bf16, As2[buf][(mi0 * 16 + mloc) * 32 + kk]);
        c1f[j] = __builtin_bit_cast(__bf16, As2[buf][((mi0 + 1) * 16 + mloc) * 32 + kk]);
      }
      acc0 = __builtin_amdgcn_wmma_f32_16x16x32_bf16(false, c0f, false, b2f,
                                                     (short)0, acc0, false, false);
      acc1 = __builtin_amdgcn_wmma_f32_16x16x32_bf16(false, c1f, false, b2f,
                                                     (short)0, acc1, false, false);
    }
  }

  // epilogue: C/D layout — VGPR i, lane L -> M = i + 8*(L>>4), N = L&15
  const int n = n0 + ni * 16 + (lane & 15);
  const float bv = bias ? bias[n] : 0.0f;
#pragma unroll
  for (int i = 0; i < 8; ++i) {
    const int mA = m0 + mi0 * 16 + i + 8 * (lane >> 4);
    if (mA < M) {
      float v = acc0[i] + bv;
      if (do_relu) v = fmaxf(v, 0.0f);
      out[(size_t)mA * N + n] = f32_to_bf16(v);
    }
    const int mB = m0 + (mi0 + 1) * 16 + i + 8 * (lane >> 4);
    if (mB < M) {
      float v = acc1[i] + bv;
      if (do_relu) v = fmaxf(v, 0.0f);
      out[(size_t)mB * N + n] = f32_to_bf16(v);
    }
  }
}

// ---------------------------------------------------------------------------
// Final classifier (256 -> 5) + softmax, f32 out. One row per thread.
// ---------------------------------------------------------------------------
__global__ __launch_bounds__(256) void fc_softmax_kernel(
    const unsigned short* __restrict__ h,  // [M,256] bf16
    const float* __restrict__ Wfc,         // [256,5]
    const float* __restrict__ bfc,         // [5]
    float* __restrict__ out, int M) {
  __shared__ float Ws[256 * 5];
  __shared__ float bs[5];
  for (int i = threadIdx.x; i < 256 * 5; i += blockDim.x) Ws[i] = Wfc[i];
  if (threadIdx.x < 5) bs[threadIdx.x] = bfc[threadIdx.x];
  __syncthreads();

  int row = blockIdx.x * blockDim.x + threadIdx.x;
  if (row >= M) return;

  float acc[5];
#pragma unroll
  for (int c = 0; c < 5; ++c) acc[c] = bs[c];
  const unsigned short* hr = h + (size_t)row * 256;
  for (int k = 0; k < 256; ++k) {
    float xv = bf16_to_f32(hr[k]);
#pragma unroll
    for (int c = 0; c < 5; ++c) acc[c] += xv * Ws[k * 5 + c];
  }
  float mx = acc[0];
#pragma unroll
  for (int c = 1; c < 5; ++c) mx = fmaxf(mx, acc[c]);
  float sum = 0.0f;
#pragma unroll
  for (int c = 0; c < 5; ++c) { acc[c] = __expf(acc[c] - mx); sum += acc[c]; }
  float inv = 1.0f / sum;
#pragma unroll
  for (int c = 0; c < 5; ++c) out[(size_t)row * 5 + c] = acc[c] * inv;
}

// ---------------------------------------------------------------------------
// Host-side orchestration
// ---------------------------------------------------------------------------
static inline size_t align256(size_t v) { return (v + 255) & ~(size_t)255; }

extern "C" void kernel_launch(void* const* d_in, const int* in_sizes, int n_in,
                              void* d_out, int out_size, void* d_ws, size_t ws_size,
                              hipStream_t stream) {
  (void)n_in; (void)out_size; (void)ws_size;
  constexpr int D = 256, H1 = 512, H2 = 256, NH = 3;

  const float* x   = (const float*)d_in[0];
  const int*   ei  = (const int*)d_in[1];     // int32 (JAX x64 disabled)
  const float* W1l = (const float*)d_in[2];
  const float* b1  = (const float*)d_in[3];
  const float* W1r = (const float*)d_in[4];
  const float* W2l = (const float*)d_in[5];
  const float* b2  = (const float*)d_in[6];
  const float* W2r = (const float*)d_in[7];
  const float* Wh  = (const float*)d_in[8];
  const float* bh  = (const float*)d_in[9];
  const float* Wfc = (const float*)d_in[10];
  const float* bfc = (const float*)d_in[11];

  const int Nn = in_sizes[0] / D;             // 50000
  const int E  = in_sizes[1] / 2;             // 800000
  const int* src = ei;
  const int* dst = ei + E;

  char* ws = (char*)d_ws;
  // Workspace layout (region A is reused across phases)
  const size_t off_deg = 0;                                   // Nn f32
  const size_t off_A   = align256((size_t)Nn * 4);            // Nn*2048 B pool
  const size_t off_agg1  = off_A;                             // Nn*256 f32
  const size_t off_x16   = off_A + (size_t)Nn * 1024;         // Nn*256 bf16
  const size_t off_g16   = off_x16 + (size_t)Nn * 512;        // Nn*256 bf16
  const size_t off_agg2  = off_A;                             // Nn*512 f32
  const size_t off_p0    = off_A;                             // Nn*256 bf16
  const size_t off_p1    = off_A + (size_t)Nn * 512;          // Nn*256 bf16
  const size_t off_B   = off_A + (size_t)Nn * 2048;           // h1: Nn*512 bf16
  const size_t off_h1  = off_B;
  const size_t off_C   = off_B + (size_t)Nn * 1024;           // agg2 bf16
  const size_t off_a2b = off_C;
  const size_t off_W   = off_C + (size_t)Nn * 1024;
  unsigned short* w1l16 = (unsigned short*)(ws + off_W);
  unsigned short* w1r16 = w1l16 + (size_t)D * H1;
  unsigned short* w2l16 = w1r16 + (size_t)D * H1;
  unsigned short* w2r16 = w2l16 + (size_t)H1 * H2;
  unsigned short* wh16  = w2r16 + (size_t)H1 * H2;            // 3*65536

  float* deg  = (float*)(ws + off_deg);
  float* agg1 = (float*)(ws + off_agg1);
  float* agg2 = (float*)(ws + off_agg2);
  unsigned short* x16  = (unsigned short*)(ws + off_x16);
  unsigned short* g16  = (unsigned short*)(ws + off_g16);
  unsigned short* h1   = (unsigned short*)(ws + off_h1);
  unsigned short* a2b  = (unsigned short*)(ws + off_a2b);
  unsigned short* p0   = (unsigned short*)(ws + off_p0);
  unsigned short* p1   = (unsigned short*)(ws + off_p1);

  const int mt64 = (Nn + 63) / 64;            // 782 M block-tiles

  auto cvt = [&](const float* in, unsigned short* o, long long cnt) {
    int blks = (int)((cnt + 255) / 256);
    f2bf_kernel<<<blks, 256, 0, stream>>>(in, o, cnt);
  };

  // Phase 0: zero accumulators, convert weights + x to bf16
  (void)hipMemsetAsync(deg, 0, (size_t)Nn * 4, stream);
  (void)hipMemsetAsync(agg1, 0, (size_t)Nn * D * 4, stream);
  cvt(x,   x16,  (long long)Nn * D);
  cvt(W1l, w1l16, (long long)D * H1);
  cvt(W1r, w1r16, (long long)D * H1);
  cvt(W2l, w2l16, (long long)H1 * H2);
  cvt(W2r, w2r16, (long long)H1 * H2);
  cvt(Wh,  wh16,  (long long)NH * H2 * H2);

  // Phase 1: SAGEConv 1
  degree_kernel<<<(E + 255) / 256, 256, 0, stream>>>(dst, deg, E);
  {
    long long tot = (long long)E << 6;        // F=256 -> 64 float4 chunks/edge
    scatter_f32_kernel<<<(int)((tot + 255) / 256), 256, 0, stream>>>(
        x, src, dst, agg1, E, D, 6);
  }
  {
    long long tot = (long long)Nn * D;
    mean_bf16_kernel<<<(int)((tot + 255) / 256), 256, 0, stream>>>(
        agg1, deg, g16, tot, 8);
  }
  gemm_bf16_wmma<<<dim3(H1 / 64, mt64), 256, 0, stream>>>(
      g16, x16, w1l16, w1r16, b1, h1, Nn, D, H1, 1);

  // Phase 2: SAGEConv 2 (region A recycled as agg2)
  (void)hipMemsetAsync(agg2, 0, (size_t)Nn * H1 * 4, stream);
  {
    long long tot = (long long)E << 7;        // F=512 -> 128 chunks/edge
    scatter_bf16_kernel<<<(int)((tot + 255) / 256), 256, 0, stream>>>(
        h1, src, dst, agg2, E, H1, 7);
  }
  {
    long long tot = (long long)Nn * H1;
    mean_bf16_kernel<<<(int)((tot + 255) / 256), 256, 0, stream>>>(
        agg2, deg, a2b, tot, 9);
  }
  gemm_bf16_wmma<<<dim3(H2 / 64, mt64), 256, 0, stream>>>(
      a2b, h1, w2l16, w2r16, b2, p0, Nn, H1, H2, 1);

  // Phase 3: hidden layers (ping-pong p0 <-> p1)
  unsigned short* pin = p0;
  unsigned short* pout = p1;
  for (int i = 0; i < NH; ++i) {
    gemm_bf16_wmma<<<dim3(H2 / 64, mt64), 256, 0, stream>>>(
        pin, nullptr, wh16 + (size_t)i * H2 * H2, nullptr,
        bh + (size_t)i * H2, pout, Nn, H2, H2, 1);
    unsigned short* t = pin; pin = pout; pout = t;
  }
  // after 3 swaps, final activations are in `pin`

  // Phase 4: classifier + softmax
  fc_softmax_kernel<<<(Nn + 255) / 256, 256, 0, stream>>>(
      pin, Wfc, bfc, (float*)d_out, Nn);
}
